// ControlPts_deformer_88304527606181
// MI455X (gfx1250) — matrix operand order repaired and checked by reading, not compiled, not run on hardware
//
#include <hip/hip_runtime.h>
#include <hip/hip_bf16.h>

// ---------------------------------------------------------------------------
// ControlPts deformer for MI455X (gfx1250, wave32, WMMA + TDM)
// Prep   : swizzle f32 weights -> f16 WMMA B-fragments in workspace (once).
// Stage A: transforms MLP (8192 rows x {20->128, 4x 128->128, 128->6}) via
//          v_wmma_f32_16x16x32_f16; per-layer weights pulled into LDS by the
//          Tensor Data Mover (tensor_load_to_lds, TENSORcnt), double-buffered
//          so TDM fetch of layer l+1 overlaps WMMA of layer l. so3_exp at end.
// Stage B: brute-force top-6 NN over 1024 LDS-resident control points.
// Stage C: weighted rigid blend with per-batch rot/trans tables in LDS.
// ---------------------------------------------------------------------------

typedef __attribute__((ext_vector_type(16))) _Float16 v16h;
typedef __attribute__((ext_vector_type(8)))  _Float16 v8h;
typedef __attribute__((ext_vector_type(8)))  float    v8f;
typedef __attribute__((ext_vector_type(4)))  unsigned int u32x4;
typedef __attribute__((ext_vector_type(8)))  int      i32x8;
typedef __attribute__((ext_vector_type(4)))  int      i32x4;

#define DH 128
#define NVN 6
#define WAVES_A 4
#define ROWS_PER_WAVE 16
#define THRES2_INV 50.0f   // 1/(2*0.1^2)

// swizzled-weight layout (halves): L0 @0 (4096), L1..L4 @4096+16384*l (16384 ea), L5 @69632 (2048)
#define WSWZ_HALVES 71680

#if defined(__AMDGCN__) && defined(__has_builtin)
#if __has_builtin(__builtin_amdgcn_tensor_load_to_lds) && __has_builtin(__builtin_amdgcn_s_wait_tensorcnt)
#define USE_TDM 1
#endif
#endif
#ifndef USE_TDM
#define USE_TDM 0
#endif

// ---- prep: global f32 [K][N] -> global f16 fragments ------------------------
// B-fragment layout per 16x16x32 tile (wave32, contiguous k order used by this code):
//   lane L: col N = L%16 ; v16h element e holds K = (L/16)*16 + e
// Fragment-major so each lane later reads its 16 halves as two ds_load_b128.
__global__ __launch_bounds__(256)
void prep_swizzle(const float* __restrict__ W, int Kdim, int Ndim, int KT, int NT,
                  _Float16* __restrict__ dst)
{
    const int ncols = NT * 16;
    const int total = KT * 32 * ncols;
    for (int i = blockIdx.x * blockDim.x + threadIdx.x; i < total;
         i += gridDim.x * blockDim.x) {
        int k = i / ncols, n = i - (i / ncols) * ncols;
        float v = (k < Kdim && n < Ndim) ? W[k * Ndim + n] : 0.0f;
        int kt = k >> 5, kk = k & 31, nt = n >> 4, nn = n & 15;
        int lane = ((kk >> 4) << 4) | nn;
        int e    = kk & 15;
        dst[(kt * NT + nt) * 512 + lane * 16 + e] = (_Float16)v;
    }
}

// ---- TDM issue: 1-D copy of n8 8-byte elements, global -> LDS ---------------
#if USE_TDM
__device__ inline void tdm_issue(const _Float16* __restrict__ gsrc,
                                 _Float16* __restrict__ ldst, int n8)
{
    unsigned long long ga = (unsigned long long)(size_t)gsrc;        // 57-bit global addr
    unsigned int       la = (unsigned int)(size_t)(void*)ldst;       // low 32b = LDS offset
    u32x4 g0;
    g0[0] = 1u;                                     // count=1, user descriptor
    g0[1] = la;                                     // lds_addr
    g0[2] = (unsigned int)ga;                       // global_addr[31:0]
    g0[3] = (unsigned int)(ga >> 32) | 0x80000000u; // global_addr[56:32] | type=2
    i32x8 g1;
    g1[0] = 0x30000;                                // data_size=3 (8B), no multicast
    g1[1] = (n8 & 0xffff) << 16;                    // tensor_dim0[15:0]
    g1[2] = 1 << 16;                                // tensor_dim0[31:16]=0, tensor_dim1=1
    g1[3] = n8 << 16;                               // tile_dim0
    g1[4] = 1;                                      // tile_dim1=1, tile_dim2=0
    g1[5] = n8;                                     // tensor_dim0_stride[31:0]
    g1[6] = 0;
    g1[7] = 0;
    i32x4 z4 = {0, 0, 0, 0};
    i32x8 z8 = {0, 0, 0, 0, 0, 0, 0, 0};
    // clang-23 / amdgpu-toolchain form: 6 args (g0, g1, g2, g3, g4, cpol)
    __builtin_amdgcn_tensor_load_to_lds(g0, g1, z4, z4, z8, 0);
}
#else
__device__ inline void lds_copy(const _Float16* __restrict__ gsrc,
                                _Float16* __restrict__ ldst, int halves, int tid)
{
    for (int i = tid * 8; i < halves; i += 128 * 8)
        *(v8h*)(ldst + i) = *(const v8h*)(gsrc + i);
}
#endif

// ---- one MLP layer for one wave's 16-row strip ------------------------------
// A-fragment (16-bit A 16x32, contiguous k order matching B): lane L: row M = L%16,
//   element e (0..15) -> K = (L/16)*16 + e
// D layout: lane L, acc[v]: row = v + (L/16)*8, col = L%16.
template <int KT, int NT, bool LAST>
__device__ inline void wave_layer(const _Float16* __restrict__ wfrag,
                                  _Float16* __restrict__ aw,
                                  float* __restrict__ outbuf,
                                  int lane)
{
    const int r  = lane & 15;
    const int hi = (lane >> 4) * 8;

    v16h afrag[KT];
#pragma unroll
    for (int kt = 0; kt < KT; ++kt) {
        const _Float16* base = aw + r * DH + kt * 32 + hi * 2;
        v8h lo = *(const v8h*)(base);
        v8h hh = *(const v8h*)(base + 8);
        v16h a;
#pragma unroll
        for (int i = 0; i < 8; ++i) { a[i] = lo[i]; a[i + 8] = hh[i]; }
        afrag[kt] = a;
    }

#pragma unroll
    for (int nt = 0; nt < NT; ++nt) {
        v8f acc = {};
#pragma unroll
        for (int kt = 0; kt < KT; ++kt) {
            const _Float16* wb = wfrag + (kt * NT + nt) * 512 + lane * 16;
            v8h blo = *(const v8h*)(wb);
            v8h bhi = *(const v8h*)(wb + 8);
            v16h bf;
#pragma unroll
            for (int i = 0; i < 8; ++i) { bf[i] = blo[i]; bf[i + 8] = bhi[i]; }
            acc = __builtin_amdgcn_wmma_f32_16x16x32_f16(
                false, afrag[kt], false, bf, (short)0, acc, false, false);
        }
        const int col   = nt * 16 + (lane & 15);
        const int rbase = (lane >> 4) * 8;
        if (LAST) {
#pragma unroll
            for (int v = 0; v < 8; ++v)
                outbuf[(rbase + v) * 16 + (lane & 15)] = acc[v];
        } else {
#pragma unroll
            for (int v = 0; v < 8; ++v) {
                float x = acc[v];
                x = x > 0.0f ? x : 0.0f;          // ReLU
                aw[(rbase + v) * DH + col] = (_Float16)x;
            }
        }
    }
}

// ---- Stage A kernel ---------------------------------------------------------
__global__ __launch_bounds__(WAVES_A * 32)
void stageA_mlp(const _Float16* __restrict__ wswz,
                const float* __restrict__ t, const float* __restrict__ cp,
                float* __restrict__ transforms_out,
                float* __restrict__ rot_ws, float* __restrict__ trans_ws,
                int Mcp, int rowsTotal)
{
    __shared__ __align__(16) _Float16 wfrag[2][4 * 8 * 512];              // 64 KB
    __shared__ __align__(16) _Float16 act[WAVES_A][ROWS_PER_WAVE * DH];   // 16 KB
    __shared__ __align__(16) float    outbuf[WAVES_A][ROWS_PER_WAVE * 16];// 4 KB

    const int tid  = threadIdx.x;
    const int lane = tid & 31;
    const int wave = tid >> 5;
    const int rowBase = (blockIdx.x * WAVES_A + wave) * ROWS_PER_WAVE;

#if USE_TDM
    if (wave == 0) tdm_issue(wswz + 0, wfrag[0], 1024);     // L0 -> buf0
#endif

    // embed: [x, sin(1x), cos(1x), sin(2x), cos(2x)] for 4 inputs -> 20, pad 32
    if (lane < 16) {
        int row  = rowBase + lane;
        int rowc = row < rowsTotal ? row : rowsTotal - 1;
        int bi = rowc / Mcp, m = rowc - bi * Mcp;
        float x[4] = { t[bi], cp[m * 3 + 0], cp[m * 3 + 1], cp[m * 3 + 2] };
        _Float16* ar = &act[wave][lane * DH];
#pragma unroll
        for (int j = 0; j < 4; ++j) {
            ar[j]      = (_Float16)x[j];
            ar[4 + j]  = (_Float16)__sinf(x[j]);
            ar[8 + j]  = (_Float16)__cosf(x[j]);
            ar[12 + j] = (_Float16)__sinf(2.0f * x[j]);
            ar[16 + j] = (_Float16)__cosf(2.0f * x[j]);
        }
#pragma unroll
        for (int j = 20; j < 32; ++j) ar[j] = (_Float16)0.0f;
    }

    // ---- L0: 20(pad32) -> 128, weights in buf0; prefetch L1 -> buf1
#if USE_TDM
    if (wave == 0) { tdm_issue(wswz + 4096, wfrag[1], 4096);
                     __builtin_amdgcn_s_wait_tensorcnt(1); }
#else
    lds_copy(wswz + 0, wfrag[0], 4096, tid);
#endif
    __syncthreads();
    wave_layer<1, 8, false>(wfrag[0], act[wave], nullptr, lane);
    __syncthreads();

    // ---- L1 (buf1); prefetch L2 -> buf0
#if USE_TDM
    if (wave == 0) { tdm_issue(wswz + 20480, wfrag[0], 4096);
                     __builtin_amdgcn_s_wait_tensorcnt(1); }
#else
    lds_copy(wswz + 4096, wfrag[1], 16384, tid);
#endif
    __syncthreads();
    wave_layer<4, 8, false>(wfrag[1], act[wave], nullptr, lane);
    __syncthreads();

    // ---- L2 (buf0); prefetch L3 -> buf1
#if USE_TDM
    if (wave == 0) { tdm_issue(wswz + 36864, wfrag[1], 4096);
                     __builtin_amdgcn_s_wait_tensorcnt(1); }
#else
    lds_copy(wswz + 20480, wfrag[0], 16384, tid);
#endif
    __syncthreads();
    wave_layer<4, 8, false>(wfrag[0], act[wave], nullptr, lane);
    __syncthreads();

    // ---- L3 (buf1); prefetch L4 -> buf0
#if USE_TDM
    if (wave == 0) { tdm_issue(wswz + 53248, wfrag[0], 4096);
                     __builtin_amdgcn_s_wait_tensorcnt(1); }
#else
    lds_copy(wswz + 36864, wfrag[1], 16384, tid);
#endif
    __syncthreads();
    wave_layer<4, 8, false>(wfrag[1], act[wave], nullptr, lane);
    __syncthreads();

    // ---- L4 (buf0); prefetch L5 -> buf1
#if USE_TDM
    if (wave == 0) { tdm_issue(wswz + 69632, wfrag[1], 512);
                     __builtin_amdgcn_s_wait_tensorcnt(1); }
#else
    lds_copy(wswz + 53248, wfrag[0], 16384, tid);
#endif
    __syncthreads();
    wave_layer<4, 8, false>(wfrag[0], act[wave], nullptr, lane);
    __syncthreads();

    // ---- L5: 128 -> 6(pad16), buf1, no ReLU
#if USE_TDM
    if (wave == 0) __builtin_amdgcn_s_wait_tensorcnt(0);
#else
    lds_copy(wswz + 69632, wfrag[1], 2048, tid);
#endif
    __syncthreads();
    wave_layer<4, 1, true>(wfrag[1], act[wave], outbuf[wave], lane);
    __syncthreads();

    // store transforms (b,M,6)
    {
        const int col   = lane & 15;
        const int rbase = (lane >> 4) * 8;
        if (col < 6) {
#pragma unroll
            for (int v = 0; v < 8; ++v) {
                int row = rowBase + rbase + v;
                if (row < rowsTotal)
                    transforms_out[(size_t)row * 6 + col] =
                        outbuf[wave][(rbase + v) * 16 + col];
            }
        }
    }

    // so3_exp: R = I + A*K + B*(w w^T - |w|^2 I)
    if (lane < 16) {
        int row = rowBase + lane;
        if (row < rowsTotal) {
            const float* o = &outbuf[wave][lane * 16];
            float wx = o[0], wy = o[1], wz = o[2];
            float t2 = wx * wx + wy * wy + wz * wz;
            float th = sqrtf(t2 + 1e-12f);
            float A  = __sinf(th) / th;
            float B  = (1.0f - __cosf(th)) / (t2 + 1e-12f);
            float R[9];
            R[0] = 1.0f + B * (wx * wx - t2);
            R[1] = -A * wz + B * wx * wy;
            R[2] =  A * wy + B * wx * wz;
            R[3] =  A * wz + B * wy * wx;
            R[4] = 1.0f + B * (wy * wy - t2);
            R[5] = -A * wx + B * wy * wz;
            R[6] = -A * wy + B * wz * wx;
            R[7] =  A * wx + B * wz * wy;
            R[8] = 1.0f + B * (wz * wz - t2);
#pragma unroll
            for (int j = 0; j < 9; ++j) rot_ws[(size_t)row * 9 + j] = R[j];
            trans_ws[(size_t)row * 3 + 0] = o[3];
            trans_ws[(size_t)row * 3 + 1] = o[4];
            trans_ws[(size_t)row * 3 + 2] = o[5];
        }
    }
}

// ---- Stage B: top-6 nearest control points + softmax weights ----------------
__global__ __launch_bounds__(256)
void stageB_knn(const float* __restrict__ p, const float* __restrict__ cp,
                int n, int Mcp,
                int* __restrict__ vn_idx, float* __restrict__ vn_w)
{
    __shared__ float cps[1024 * 3];
    for (int i = threadIdx.x; i < Mcp * 3; i += blockDim.x) cps[i] = cp[i];
    __syncthreads();

    int i = blockIdx.x * blockDim.x + threadIdx.x;
    if (i >= n) return;
    float px = p[i * 3], py = p[i * 3 + 1], pz = p[i * 3 + 2];

    float bd[NVN]; int bi[NVN];
#pragma unroll
    for (int k = 0; k < NVN; ++k) { bd[k] = 3.4e38f; bi[k] = 0; }

    for (int m = 0; m < Mcp; ++m) {
        float dx = px - cps[m * 3], dy = py - cps[m * 3 + 1], dz = pz - cps[m * 3 + 2];
        float d2 = dx * dx + dy * dy + dz * dz;
        if (d2 < bd[NVN - 1]) {
            bd[NVN - 1] = d2; bi[NVN - 1] = m;
#pragma unroll
            for (int j = NVN - 2; j >= 0; --j) {
                if (bd[j + 1] < bd[j]) {
                    float td = bd[j]; bd[j] = bd[j + 1]; bd[j + 1] = td;
                    int   ti = bi[j]; bi[j] = bi[j + 1]; bi[j + 1] = ti;
                }
            }
        }
    }

    float e[NVN], s = 0.0f;
#pragma unroll
    for (int k = 0; k < NVN; ++k) { e[k] = __expf(-(bd[k] - bd[0]) * THRES2_INV); s += e[k]; }
    float inv = 1.0f / s;
#pragma unroll
    for (int k = 0; k < NVN; ++k) {
        vn_idx[(size_t)i * NVN + k] = bi[k];
        vn_w  [(size_t)i * NVN + k] = e[k] * inv;
    }
}

// ---- Stage C: weighted rigid blend ------------------------------------------
__global__ __launch_bounds__(256)
void stageC_deform(const float* __restrict__ p,
                   const float* __restrict__ rot_ws, const float* __restrict__ trans_ws,
                   const int* __restrict__ vn_idx, const float* __restrict__ vn_w,
                   float* __restrict__ out, int n, int Mcp)
{
    __shared__ float Rs[1024 * 9];
    __shared__ float Ts[1024 * 3];
    const int bb = blockIdx.y;
    for (int i = threadIdx.x; i < Mcp * 9; i += blockDim.x)
        Rs[i] = rot_ws[(size_t)bb * Mcp * 9 + i];
    for (int i = threadIdx.x; i < Mcp * 3; i += blockDim.x)
        Ts[i] = trans_ws[(size_t)bb * Mcp * 3 + i];
    __syncthreads();

    int i = blockIdx.x * blockDim.x + threadIdx.x;
    if (i >= n) return;
    float px = p[i * 3], py = p[i * 3 + 1], pz = p[i * 3 + 2];
    float ax = 0.0f, ay = 0.0f, az = 0.0f;
#pragma unroll
    for (int k = 0; k < NVN; ++k) {
        int   m = vn_idx[(size_t)i * NVN + k];
        float w = vn_w  [(size_t)i * NVN + k];
        const float* R = &Rs[m * 9];
        const float* T = &Ts[m * 3];
        ax += w * (R[0] * px + R[1] * py + R[2] * pz + T[0]);
        ay += w * (R[3] * px + R[4] * py + R[5] * pz + T[1]);
        az += w * (R[6] * px + R[7] * py + R[8] * pz + T[2]);
    }
    size_t o = ((size_t)bb * n + i) * 3;
    out[o] = ax; out[o + 1] = ay; out[o + 2] = az;
}

// ---------------------------------------------------------------------------
extern "C" void kernel_launch(void* const* d_in, const int* in_sizes, int n_in,
                              void* d_out, int out_size, void* d_ws, size_t ws_size,
                              hipStream_t stream)
{
    const float* p  = (const float*)d_in[0];
    const float* t  = (const float*)d_in[1];
    const float* cp = (const float*)d_in[2];
    const float* W0 = (const float*)d_in[3];
    const float* W1 = (const float*)d_in[4];
    const float* W2 = (const float*)d_in[5];
    const float* W3 = (const float*)d_in[6];
    const float* W4 = (const float*)d_in[7];
    const float* W5 = (const float*)d_in[8];

    const int n   = in_sizes[0] / 3;   // 200000
    const int b   = in_sizes[1];       // 8
    const int Mcp = in_sizes[2] / 3;   // 1024
    const int rows = b * Mcp;          // 8192

    float* out = (float*)d_out;
    float* transforms_out = out + (size_t)b * n * 3;

    // workspace: wswz (71680 f16) | rot (rows*9 f32) | trans (rows*3 f32)
    //          | vn_idx (n*6 i32) | vn_w (n*6 f32)
    _Float16* wswz = (_Float16*)d_ws;
    float* rot_ws   = (float*)((char*)d_ws + (size_t)WSWZ_HALVES * 2);
    float* trans_ws = rot_ws + (size_t)rows * 9;
    int*   vn_idx   = (int*)(trans_ws + (size_t)rows * 3);
    float* vn_w     = (float*)(vn_idx + (size_t)n * NVN);

    // prep: swizzle all weight layers into fragment layout
    prep_swizzle<<<16, 256, 0, stream>>>(W0, 20,  DH, 1, 8, wswz + 0);
    prep_swizzle<<<16, 256, 0, stream>>>(W1, DH,  DH, 4, 8, wswz + 4096);
    prep_swizzle<<<16, 256, 0, stream>>>(W2, DH,  DH, 4, 8, wswz + 20480);
    prep_swizzle<<<16, 256, 0, stream>>>(W3, DH,  DH, 4, 8, wswz + 36864);
    prep_swizzle<<<16, 256, 0, stream>>>(W4, DH,  DH, 4, 8, wswz + 53248);
    prep_swizzle<<<16, 256, 0, stream>>>(W5, DH,  6,  4, 1, wswz + 69632);

    const int rowsPerBlock = WAVES_A * ROWS_PER_WAVE;  // 64
    const int gridA = (rows + rowsPerBlock - 1) / rowsPerBlock;
    stageA_mlp<<<gridA, WAVES_A * 32, 0, stream>>>(
        wswz, t, cp, transforms_out, rot_ws, trans_ws, Mcp, rows);

    stageB_knn<<<(n + 255) / 256, 256, 0, stream>>>(p, cp, n, Mcp, vn_idx, vn_w);

    stageC_deform<<<dim3((n + 255) / 256, b), 256, 0, stream>>>(
        p, rot_ws, trans_ws, vn_idx, vn_w, out, n, Mcp);
}
